// Attention_10840497455329
// MI455X (gfx1250) — matrix operand order repaired and checked by reading, not compiled, not run on hardware
//
#include <hip/hip_runtime.h>
#include <stddef.h>
#include <stdint.h>

// ---------------- problem constants ----------------
constexpr int S_   = 2048;
constexpr int DIM_ = 4096;
constexpr int H_   = 32;
constexpr int HKV_ = 8;
constexpr int HD_  = 128;
constexpr int REP_ = H_ / HKV_;       // 4
constexpr int HALF_ = HD_ / 2;        // 64
constexpr float SCALE_ = 0.08838834764831845f;  // 1/sqrt(128)
constexpr float NEG_ = -1000000000.0f;

// ---------------- vector types / helpers ----------------
typedef __bf16         bf16x16 __attribute__((ext_vector_type(16)));
typedef float          f32x8   __attribute__((ext_vector_type(8)));
typedef unsigned short u16x8   __attribute__((ext_vector_type(8)));
typedef int            i32x4   __attribute__((ext_vector_type(4)));

union FragU { u16x8 h[2]; bf16x16 b; };

__device__ __forceinline__ float bf2f(unsigned short u) {
    union { unsigned int i; float f; } v;
    v.i = ((unsigned int)u) << 16;
    return v.f;
}
__device__ __forceinline__ unsigned short f2bf(float f) {
    union { float f; unsigned int i; } v;
    v.f = f;
    unsigned int r = v.i + 0x7FFFu + ((v.i >> 16) & 1u);   // RNE
    return (unsigned short)(r >> 16);
}

// A fragment: 16x32 bf16, row m = lane&15, kSel = lane>=16, K = 16*kSel + i (16 contiguous elems).
__device__ __forceinline__ bf16x16 load_frag_a(const unsigned short* rowPtr, int k0, int kSel) {
    FragU f;
    f.h[0] = *(const u16x8*)(rowPtr + k0 + 16 * kSel);
    f.h[1] = *(const u16x8*)(rowPtr + k0 + 16 * kSel + 8);
    return f.b;
}
// B fragment: 32x16 bf16, col n = lane&15, K = 16*kSel + i (16 contiguous elems)
__device__ __forceinline__ bf16x16 load_frag_b(const unsigned short* p) {
    FragU f;
    f.h[0] = *(const u16x8*)(p);
    f.h[1] = *(const u16x8*)(p + 8);
    return f.b;
}

#define WMMA_BF16(a, b, c) \
    __builtin_amdgcn_wmma_f32_16x16x32_bf16(false, (a), false, (b), (short)0, (c), false, false)

// ---------------- async global -> LDS copy (gfx1250 ASYNCcnt path) ----------------
#if defined(__HIP_DEVICE_COMPILE__)
#if __has_builtin(__builtin_amdgcn_global_load_async_to_lds_b128)
#define HAVE_ASYNC_LDS 1
#endif
#endif
#ifndef HAVE_ASYNC_LDS
#define HAVE_ASYNC_LDS 0
#endif

__device__ __forceinline__ void copy16_g2l(const unsigned short* g, unsigned short* l) {
#if HAVE_ASYNC_LDS
    typedef __attribute__((address_space(1))) i32x4* gv4p;   // global int4*
    typedef __attribute__((address_space(3))) i32x4* lv4p;   // LDS int4*
    __builtin_amdgcn_global_load_async_to_lds_b128(
        (gv4p)(uintptr_t)g,
        (lv4p)(unsigned int)(uintptr_t)l, 0, 0);
#else
    *(u16x8*)l = *(const u16x8*)g;
#endif
}
__device__ __forceinline__ void wait_async_copies() {
#if HAVE_ASYNC_LDS
    asm volatile("s_wait_asynccnt 0x0" ::: "memory");
#endif
}

// ---------------- bf16 GEMM: Out[M,N] = X[M,K] * W[N,K]^T (+ bias) ----------------
// block = 128 threads (4 waves). Block tile 64(M) x 128(N); wave tile 16 x 128.
// B tile (128 rows x 32 K) is staged in LDS (double-buffered, async copy) and
// shared by all 4 waves -> 4x less weight traffic from L2.
constexpr int GN_ = 128;   // block N tile
__global__ void __launch_bounds__(128)
gemm_bf16(const unsigned short* __restrict__ X,
          const unsigned short* __restrict__ W,
          const unsigned short* __restrict__ bias,
          unsigned short* __restrict__ Out,
          int N, int Kdim)
{
    __shared__ __align__(16) unsigned short ldsB[2][GN_ * 32];

    const int tid  = threadIdx.x;
    const int wave = tid >> 5;
    const int lane = tid & 31;
    const int m0   = blockIdx.y * 64 + wave * 16;
    const int n0   = blockIdx.x * GN_;
    const int mn   = lane & 15;
    const int kSel = lane >> 4;

    const unsigned short* xrow = X + (size_t)(m0 + mn) * Kdim;
    const unsigned short* wrow = W + (size_t)(n0 + tid) * Kdim;  // thread t copies W row n0+t
    unsigned short* l0 = &ldsB[0][tid * 32];
    unsigned short* l1 = &ldsB[1][tid * 32];

    auto stageB = [&](unsigned short* ldst, int k0) {
#pragma unroll
        for (int j = 0; j < 4; ++j)
            copy16_g2l(wrow + k0 + j * 8, ldst + j * 8);
    };

    f32x8 acc[8];
#pragma unroll
    for (int t = 0; t < 8; ++t) acc[t] = f32x8{};

    stageB(l0, 0);
    wait_async_copies();
    __syncthreads();

    int buf = 0;
    for (int k0 = 0; k0 < Kdim; k0 += 32) {
        if (k0 + 32 < Kdim) stageB(buf ? l0 : l1, k0 + 32);   // prefetch next tile
        bf16x16 a = load_frag_a(xrow, k0, kSel);
        const unsigned short* bbase = &ldsB[buf][16 * kSel];
#pragma unroll
        for (int t = 0; t < 8; ++t) {
            bf16x16 b = load_frag_b(bbase + (t * 16 + mn) * 32);
            acc[t] = WMMA_BF16(a, b, acc[t]);
        }
        wait_async_copies();
        __syncthreads();
        buf ^= 1;
    }

#pragma unroll
    for (int t = 0; t < 8; ++t) {
        const int col = n0 + t * 16 + mn;
        const float bv = bias ? bf2f(bias[col]) : 0.0f;
#pragma unroll
        for (int r = 0; r < 8; ++r) {
            const int row = m0 + r + 8 * kSel;
            Out[(size_t)row * N + col] = f2bf(acc[t][r] + bv);
        }
    }
}

// ---------------- RoPE on Q [S,H,HD] and K [S,HKV,HD] ----------------
__global__ void rope_kernel(unsigned short* __restrict__ Q,
                            unsigned short* __restrict__ Kt,
                            const float* __restrict__ fcos,
                            const float* __restrict__ fsin)
{
    const int tid = blockIdx.x * blockDim.x + threadIdx.x;
    const int total = S_ * (H_ + HKV_) * HALF_;
    if (tid >= total) return;
    const int i  = tid % HALF_;
    const int he = (tid / HALF_) % (H_ + HKV_);
    const int sq = tid / (HALF_ * (H_ + HKV_));
    unsigned short* p = (he < H_)
        ? Q  + ((size_t)sq * H_   +  he)       * HD_
        : Kt + ((size_t)sq * HKV_ + (he - H_)) * HD_;
    const float a  = bf2f(p[i]);
    const float b  = bf2f(p[i + HALF_]);
    const float cc = fcos[sq * HALF_ + i];
    const float ss = fsin[sq * HALF_ + i];
    p[i]         = f2bf(a * cc - b * ss);
    p[i + HALF_] = f2bf(a * ss + b * cc);
}

// ---------------- V transpose: [S,HKV,HD] -> Vt[HKV,HD,S] ----------------
__global__ void transpose_v(const unsigned short* __restrict__ V,
                            unsigned short* __restrict__ Vt)
{
    const int tid = blockIdx.x * blockDim.x + threadIdx.x;
    const int total = S_ * HKV_ * HD_;
    if (tid >= total) return;
    const int d  = tid % HD_;
    const int kv = (tid / HD_) % HKV_;
    const int sq = tid / (HD_ * HKV_);
    Vt[((size_t)kv * HD_ + d) * S_ + sq] = V[tid];
}

// ---------------- flash attention ----------------
// grid = (H, S/64); block = 128 (4 waves); wave -> one head x one 16-row query tile.
__global__ void __launch_bounds__(128)
attn_kernel(const unsigned short* __restrict__ Q,
            const unsigned short* __restrict__ K,
            const unsigned short* __restrict__ Vt,
            unsigned short* __restrict__ O)
{
    __shared__ unsigned short ldsP[4][16 * 32];   // per-wave P tile (16x32 bf16)

    const int wave = threadIdx.x >> 5;
    const int lane = threadIdx.x & 31;
    const int h    = blockIdx.x;
    const int q0   = (blockIdx.y * 4 + wave) * 16;
    const int hkv  = h / REP_;
    const int mn   = lane & 15;
    const int kSel = lane >> 4;

    const unsigned short* qrow = Q + ((size_t)(q0 + mn) * H_ + h) * HD_;
    bf16x16 aq[4];
#pragma unroll
    for (int kc = 0; kc < 4; ++kc) aq[kc] = load_frag_a(qrow, kc * 32, kSel);

    f32x8 accO[8];
    float mOld[8], lSum[8], alpha[8];
#pragma unroll
    for (int dt = 0; dt < 8; ++dt) accO[dt] = f32x8{};
#pragma unroll
    for (int r = 0; r < 8; ++r) { mOld[r] = -1e30f; lSum[r] = 0.0f; }

    unsigned short* myP = &ldsP[wave][0];
    const int kbMax = (q0 + 15) / 32;

    for (int kb = 0; kb <= kbMax; ++kb) {
        const int keyBase = kb * 32;

        // ---- scores: S(16x32) = Q(16x128) . K_block^T(128x32) ----
        f32x8 sc0 = {}, sc1 = {};
        const unsigned short* kr0 =
            K + ((size_t)(keyBase + mn)      * HKV_ + hkv) * HD_ + 16 * kSel;
        const unsigned short* kr1 =
            K + ((size_t)(keyBase + 16 + mn) * HKV_ + hkv) * HD_ + 16 * kSel;
#pragma unroll
        for (int kc = 0; kc < 4; ++kc) {
            sc0 = WMMA_BF16(aq[kc], load_frag_b(kr0 + kc * 32), sc0);
            sc1 = WMMA_BF16(aq[kc], load_frag_b(kr1 + kc * 32), sc1);
        }

        // ---- online softmax (rows live in one 16-lane half-wave) ----
#pragma unroll
        for (int r = 0; r < 8; ++r) {
            const int qg = q0 + r + 8 * kSel;
            float s0 = sc0[r] * SCALE_ + ((keyBase + mn)      <= qg ? 0.0f : NEG_);
            float s1 = sc1[r] * SCALE_ + ((keyBase + 16 + mn) <= qg ? 0.0f : NEG_);
            float mx = fmaxf(s0, s1);
#pragma unroll
            for (int off = 1; off < 16; off <<= 1)
                mx = fmaxf(mx, __shfl_xor(mx, off, 32));
            const float mNew = fmaxf(mOld[r], mx);
            const float al = __expf(mOld[r] - mNew);
            const float p0 = __expf(s0 - mNew);
            const float p1 = __expf(s1 - mNew);
            float rs = p0 + p1;
#pragma unroll
            for (int off = 1; off < 16; off <<= 1)
                rs += __shfl_xor(rs, off, 32);
            lSum[r] = lSum[r] * al + rs;
            mOld[r] = mNew;
            alpha[r] = al;
            const int row = r + 8 * kSel;
            myP[row * 32 + mn]      = f2bf(p0);
            myP[row * 32 + 16 + mn] = f2bf(p1);
        }

#pragma unroll
        for (int dt = 0; dt < 8; ++dt)
#pragma unroll
            for (int r = 0; r < 8; ++r) accO[dt][r] *= alpha[r];

        asm volatile("s_wait_dscnt 0x0" ::: "memory");
        bf16x16 ap = load_frag_a(myP + mn * 32, 0, kSel);

        // ---- accO(16x128) += P(16x32) . V_block(32x128), Vt is [hkv][d][s] ----
#pragma unroll
        for (int dt = 0; dt < 8; ++dt) {
            const unsigned short* vr =
                Vt + ((size_t)hkv * HD_ + dt * 16 + mn) * S_ + keyBase + 16 * kSel;
            accO[dt] = WMMA_BF16(ap, load_frag_b(vr), accO[dt]);
        }
    }

#pragma unroll
    for (int dt = 0; dt < 8; ++dt) {
#pragma unroll
        for (int r = 0; r < 8; ++r) {
            const int row = q0 + r + 8 * kSel;
            const int col = h * HD_ + dt * 16 + mn;
            O[(size_t)row * (H_ * HD_) + col] = f2bf(accO[dt][r] / lSum[r]);
        }
    }
}

// ---------------- launcher ----------------
extern "C" void kernel_launch(void* const* d_in, const int* in_sizes, int n_in,
                              void* d_out, int out_size, void* d_ws, size_t ws_size,
                              hipStream_t stream)
{
    const unsigned short* x    = (const unsigned short*)d_in[0];
    const float*          fcos = (const float*)d_in[1];
    const float*          fsin = (const float*)d_in[2];
    // d_in[3] = mask (recomputed analytically), d_in[4] = positions (implicit)
    const unsigned short* wq = (const unsigned short*)d_in[5];
    const unsigned short* bq = (const unsigned short*)d_in[6];
    const unsigned short* wk = (const unsigned short*)d_in[7];
    const unsigned short* bk = (const unsigned short*)d_in[8];
    const unsigned short* wv = (const unsigned short*)d_in[9];
    const unsigned short* bv = (const unsigned short*)d_in[10];
    const unsigned short* wo = (const unsigned short*)d_in[11];
    unsigned short* out = (unsigned short*)d_out;

    unsigned short* Qb = (unsigned short*)d_ws;                 // 16 MB
    unsigned short* Kb = Qb + (size_t)S_ * H_   * HD_;          //  4 MB
    unsigned short* Vb = Kb + (size_t)S_ * HKV_ * HD_;          //  4 MB
    unsigned short* Vt = Vb + (size_t)S_ * HKV_ * HD_;          //  4 MB
    unsigned short* Ab = Vt + (size_t)S_ * HKV_ * HD_;          // 16 MB

    // QKV projections (WMMA GEMM, LDS-staged weights)
    gemm_bf16<<<dim3((H_   * HD_) / GN_, S_ / 64), 128, 0, stream>>>(x, wq, bq, Qb, H_   * HD_, DIM_);
    gemm_bf16<<<dim3((HKV_ * HD_) / GN_, S_ / 64), 128, 0, stream>>>(x, wk, bk, Kb, HKV_ * HD_, DIM_);
    gemm_bf16<<<dim3((HKV_ * HD_) / GN_, S_ / 64), 128, 0, stream>>>(x, wv, bv, Vb, HKV_ * HD_, DIM_);

    // RoPE on Q and K
    {
        const int total = S_ * (H_ + HKV_) * HALF_;
        rope_kernel<<<(total + 255) / 256, 256, 0, stream>>>(Qb, Kb, fcos, fsin);
    }

    // V transpose for contiguous PV B-fragments
    {
        const int total = S_ * HKV_ * HD_;
        transpose_v<<<(total + 255) / 256, 256, 0, stream>>>(Vb, Vt);
    }

    // flash attention
    attn_kernel<<<dim3(H_, S_ / 64), 128, 0, stream>>>(Qb, Kb, Vt, Ab);

    // output projection (WMMA GEMM, no bias)
    gemm_bf16<<<dim3(DIM_ / GN_, S_ / 64), 128, 0, stream>>>(Ab, wo, nullptr, out, DIM_, DIM_);
}
